// CTRNN_71889162600623
// MI455X (gfx1250) — matrix operand order repaired and checked by reading, not compiled, not run on hardware
//
#include <hip/hip_runtime.h>

typedef __attribute__((ext_vector_type(16))) __bf16          v16bf;
typedef __attribute__((ext_vector_type(16))) float           v16f;
typedef __attribute__((ext_vector_type(8)))  unsigned        v8u;
typedef __attribute__((ext_vector_type(4)))  unsigned        v4u;
typedef __attribute__((ext_vector_type(8)))  float           v8f;
typedef __attribute__((ext_vector_type(4)))  float           v4f;

#define S_STEPS 1024
#define BATCH   128
#define INP     512
#define HID     512
#define BHH     (BATCH * HID)
#define NWG_REC 16

#define ALPHA_C 0.2f
#define OMA_C   0.8f
#define SIGMA_C 0.15811388300841897f   // sqrt(2/alpha)*0.05

// ---------------------------------------------------------------------------
// float -> bf16 via native fptrunc (RNE). Backend picks v_cvt_pk_bf16_f32 if
// the ISA has it; otherwise expands inline — never worse than manual bits.
// ---------------------------------------------------------------------------
__device__ __forceinline__ unsigned short f2bf(float f) {
    return __builtin_bit_cast(unsigned short, (__bf16)f);
}

__device__ __forceinline__ v16bf cvt16(v16f f) {
    return __builtin_convertvector(f, v16bf);
}

// load a 16-bf16 fragment from LDS with only 16B alignment required
__device__ __forceinline__ v16bf lds_frag(const unsigned short* p) {
    v4u lo = *(const v4u*)p;
    v4u hi = *(const v4u*)(p + 8);
    v8u u;
    u[0] = lo[0]; u[1] = lo[1]; u[2] = lo[2]; u[3] = lo[3];
    u[4] = hi[0]; u[5] = hi[1]; u[6] = hi[2]; u[7] = hi[3];
    return __builtin_bit_cast(v16bf, u);
}

// ---------------------------------------------------------------------------
// Phase 1: P = x @ w_in^T + b_in + b_hh + sigma*noise   (131072x512 @ 512x512)
// WG = 8 waves, macro-tile 128(M) x 64(N); w_in strip in LDS as bf16
// B-fragments; wave = 16 rows x 4 n-tiles; A loaded once per chunk, used 4x.
// ---------------------------------------------------------------------------
__global__ __launch_bounds__(256) void ctrnn_input_gemm(
    const float* __restrict__ x, const float* __restrict__ noise,
    const float* __restrict__ w_in, const float* __restrict__ b_in,
    const float* __restrict__ b_hh, float* __restrict__ out)
{
    __shared__ __align__(32) unsigned short ldsB[16 * 64 * 32];  // [c][n64][k32]

    const int lane = threadIdx.x & 31;
    const int wave = threadIdx.x >> 5;
    const int half = lane >> 4;
    const int l16  = lane & 15;

    const int ng     = blockIdx.x & 7;   // column group (64 cols)
    const int mgroup = blockIdx.x >> 3;  // 1024 groups of 128 rows
    const int n0g = ng * 64;
    const int m0  = (mgroup * 8 + wave) * 16;

    for (int i = threadIdx.x; i < 64 * 512; i += 256) {
        const int nl = i >> 9;
        const int k  = i & 511;
        ldsB[(k >> 5) * 2048 + nl * 32 + (k & 31)] =
            f2bf(w_in[(size_t)(n0g + nl) * INP + k]);
    }
    __syncthreads();

    const float* arow = x + (size_t)(m0 + l16) * INP;
    v8f acc0 = {}, acc1 = {}, acc2 = {}, acc3 = {};

    for (int c = 0; c < INP / 32; ++c) {
        const int k0 = c * 32;
        // A 16x32 bf16 layout: e -> k0+16*half+e
        const v4f a0 = *(const v4f*)(arow + k0 + (half << 4));
        const v4f a1 = *(const v4f*)(arow + k0 + (half << 4) + 4);
        const v4f a2 = *(const v4f*)(arow + k0 + (half << 4) + 8);
        const v4f a3 = *(const v4f*)(arow + k0 + (half << 4) + 12);
        v16f af;
        af[0]  = a0[0]; af[1]  = a0[1]; af[2]  = a0[2]; af[3]  = a0[3];
        af[4]  = a1[0]; af[5]  = a1[1]; af[6]  = a1[2]; af[7]  = a1[3];
        af[8]  = a2[0]; af[9]  = a2[1]; af[10] = a2[2]; af[11] = a2[3];
        af[12] = a3[0]; af[13] = a3[1]; af[14] = a3[2]; af[15] = a3[3];
        const v16bf a = cvt16(af);

        const unsigned short* bbase = &ldsB[c * 2048 + l16 * 32 + (half << 4)];
        acc0 = __builtin_amdgcn_wmma_f32_16x16x32_bf16(false, a, false,
                   lds_frag(bbase +    0), (short)0, acc0, false, false);
        acc1 = __builtin_amdgcn_wmma_f32_16x16x32_bf16(false, a, false,
                   lds_frag(bbase +  512), (short)0, acc1, false, false);
        acc2 = __builtin_amdgcn_wmma_f32_16x16x32_bf16(false, a, false,
                   lds_frag(bbase + 1024), (short)0, acc2, false, false);
        acc3 = __builtin_amdgcn_wmma_f32_16x16x32_bf16(false, a, false,
                   lds_frag(bbase + 1536), (short)0, acc3, false, false);
    }

    v8f* accs[4] = { &acc0, &acc1, &acc2, &acc3 };
#pragma unroll
    for (int t = 0; t < 4; ++t) {
        const int   n    = n0g + t * 16 + l16;
        const float bias = b_in[n] + b_hh[n];
#pragma unroll
        for (int r = 0; r < 8; ++r) {
            const int m = m0 + r + (half << 3);
            const size_t idx = (size_t)m * HID + n;
            out[idx] = (*accs[t])[r] + bias + SIGMA_C * noise[idx];
        }
    }
}

// ---------------------------------------------------------------------------
// Grid-wide software barrier (counter + generation)
// ---------------------------------------------------------------------------
__device__ __forceinline__ void grid_barrier(unsigned* cnt, unsigned* gen,
                                             unsigned nwg)
{
    __threadfence();
    __syncthreads();
    if (threadIdx.x == 0) {
        const unsigned g = __hip_atomic_load(gen, __ATOMIC_RELAXED,
                                             __HIP_MEMORY_SCOPE_AGENT);
        const unsigned arrived = __hip_atomic_fetch_add(
            cnt, 1u, __ATOMIC_ACQ_REL, __HIP_MEMORY_SCOPE_AGENT);
        if (arrived == nwg - 1u) {
            __hip_atomic_store(cnt, 0u, __ATOMIC_RELAXED,
                               __HIP_MEMORY_SCOPE_AGENT);
            __hip_atomic_fetch_add(gen, 1u, __ATOMIC_ACQ_REL,
                                   __HIP_MEMORY_SCOPE_AGENT);
        } else {
            while (__hip_atomic_load(gen, __ATOMIC_ACQUIRE,
                                     __HIP_MEMORY_SCOPE_AGENT) == g) {
                __builtin_amdgcn_s_sleep(1);
            }
        }
    }
    __syncthreads();
}

// ---------------------------------------------------------------------------
// Phase 2: persistent recurrence.  h_s = 0.8 h + 0.2 relu(P[s] + h @ w_hh^T)
// 16 WGs = 8 b-tiles x 2 g-halves. WG: 16 rows x 256 cols; its 256 w_hh rows
// live as bf16 fragments in 256 KB LDS for all steps. Per step the WG's 16
// h-rows (bf16, double-buffered in d_ws) are staged once into 16 KB LDS and
// shared by all 8 waves; f32 carry state stays in d_out.
// ---------------------------------------------------------------------------
template <bool USE_HB>
__global__ __launch_bounds__(256) void ctrnn_recurrence(
    const float* __restrict__ w_hh, const float* __restrict__ h0,
    float* __restrict__ out, unsigned* __restrict__ bar,
    unsigned short* __restrict__ hbuf0, unsigned short* __restrict__ hbuf1)
{
    extern __shared__ unsigned short ldsB[]; // [c][n256][k32]: c*8192+n*32+k
    unsigned short* ldsA = ldsB + 16 * 8192; // 16 rows x 512 bf16 (16 KB)

    const int lane = threadIdx.x & 31;
    const int wave = threadIdx.x >> 5;
    const int half = lane >> 4;
    const int l16  = lane & 15;

    const int gh    = blockIdx.x & 1;    // column half (256 cols)
    const int btile = blockIdx.x >> 1;   // 0..7
    const int b0    = btile * 16;
    const int gbase = gh * 256;
    const int nloc  = wave * 32;         // wave's 2 n-tiles inside the half

    // stage w_hh rows [gbase, gbase+256) as bf16 fragments (once)
    for (int i = threadIdx.x; i < 256 * 512; i += 256) {
        const int nl = i >> 9;
        const int k  = i & 511;
        ldsB[(k >> 5) * 8192 + nl * 32 + (k & 31)] =
            f2bf(w_hh[(size_t)(gbase + nl) * HID + k]);
    }
    // convert h0 -> hbuf1 (h_{-1}) cooperatively
    if (USE_HB) {
        for (int i = blockIdx.x * 256 + threadIdx.x; i < BHH; i += NWG_REC * 256)
            hbuf1[i] = f2bf(h0[i]);
    }
    __syncthreads();

    unsigned* cnt = bar;
    unsigned* gen = bar + 32;
    if (USE_HB) grid_barrier(cnt, gen, NWG_REC);

    for (int s = 0; s < S_STEPS; ++s) {
        const float* hsrcf = (s == 0) ? h0 : out + (size_t)(s - 1) * BHH;
        float*       dst   = out + (size_t)s * BHH;
        const unsigned short* hsb = (s & 1) ? hbuf0 : hbuf1;  // holds h_{s-1}
        unsigned short*       hdb = (s & 1) ? hbuf1 : hbuf0;  // receives h_s

        // stage this WG's 16 h-rows into LDS (shared by all 8 waves)
        if (USE_HB) {
            const v4u* src = (const v4u*)(hsb + (size_t)b0 * HID);
            v4u*       dls = (v4u*)ldsA;
#pragma unroll
            for (int i = 0; i < 4; ++i)
                dls[threadIdx.x + 256 * i] = src[threadIdx.x + 256 * i];
        } else {
            const float* src = hsrcf + (size_t)b0 * HID;
            for (int i = threadIdx.x; i < 16 * 512; i += 256)
                ldsA[i] = f2bf(src[i]);
        }
        __syncthreads();

        v8f acc0 = {}, acc1 = {};
        const unsigned short* ar = ldsA + l16 * 512;
#pragma unroll 4
        for (int c = 0; c < HID / 32; ++c) {
            const v4u r0 = *(const v4u*)(ar + c * 32 + (half << 4));
            const v4u r1 = *(const v4u*)(ar + c * 32 + (half << 4) + 8);
            v8u au;
            au[0] = r0[0]; au[1] = r0[1]; au[2] = r0[2]; au[3] = r0[3];
            au[4] = r1[0]; au[5] = r1[1]; au[6] = r1[2]; au[7] = r1[3];
            const v16bf a = __builtin_bit_cast(v16bf, au);
            const unsigned short* bbase =
                &ldsB[c * 8192 + (nloc + l16) * 32 + (half << 4)];
            acc0 = __builtin_amdgcn_wmma_f32_16x16x32_bf16(false, a, false,
                       lds_frag(bbase), (short)0, acc0, false, false);
            acc1 = __builtin_amdgcn_wmma_f32_16x16x32_bf16(false, a, false,
                       lds_frag(bbase + 512), (short)0, acc1, false, false);
        }

        v8f* accs[2] = { &acc0, &acc1 };
#pragma unroll
        for (int t = 0; t < 2; ++t) {
            const int n = gbase + nloc + t * 16 + l16;
#pragma unroll
            for (int r = 0; r < 8; ++r) {
                const int m = b0 + r + (half << 3);
                const size_t idx = (size_t)m * HID + n;
                const float pre = (*accs[t])[r] + dst[idx];  // dst holds P[s]
                const float hp  = hsrcf[idx];
                const float hn  = OMA_C * hp +
                                  ALPHA_C * (pre > 0.0f ? pre : 0.0f);
                dst[idx] = hn;
                if (USE_HB) hdb[idx] = f2bf(hn);
            }
        }

        if (s != S_STEPS - 1) {
            // hide part of next step's P read behind the barrier spin
            const float* nxt = dst + BHH;
#pragma unroll
            for (int r = 0; r < 8; ++r)
                __builtin_prefetch(
                    nxt + (size_t)(b0 + r + (half << 3)) * HID + gbase + nloc + l16,
                    0, 1);
            grid_barrier(cnt, gen, NWG_REC);
        }
    }
}

// ---------------------------------------------------------------------------
extern "C" void kernel_launch(void* const* d_in, const int* in_sizes, int n_in,
                              void* d_out, int out_size, void* d_ws,
                              size_t ws_size, hipStream_t stream)
{
    const float* x     = (const float*)d_in[0];
    const float* noise = (const float*)d_in[1];
    const float* w_in  = (const float*)d_in[2];
    const float* b_in  = (const float*)d_in[3];
    const float* w_hh  = (const float*)d_in[4];
    const float* b_hh  = (const float*)d_in[5];
    const float* h0    = (const float*)d_in[6];
    float* out = (float*)d_out;

    // ws layout: [0,256) barrier state ; [256, 256 + 2*BHH*2) bf16 h buffers
    const size_t need_hb = 256 + 2 * (size_t)BHH * sizeof(unsigned short);
    const bool   use_hb  = ws_size >= need_hb;
    unsigned short* hb0 = (unsigned short*)((char*)d_ws + 256);
    unsigned short* hb1 = hb0 + BHH;

    const size_t clr = ws_size < 256 ? ws_size : (size_t)256;
    hipMemsetAsync(d_ws, 0, clr, stream);

    dim3 block(256);
    ctrnn_input_gemm<<<dim3(1024 * 8), block, 0, stream>>>(x, noise, w_in,
                                                           b_in, b_hh, out);
    const size_t ldsBytes =
        (16 * 8192 + 16 * 512) * sizeof(unsigned short);   // 256 KB + 16 KB
    if (use_hb) {
        ctrnn_recurrence<true><<<NWG_REC, block, ldsBytes, stream>>>(
            w_hh, h0, out, (unsigned*)d_ws, hb0, hb1);
    } else {
        ctrnn_recurrence<false><<<NWG_REC, block, ldsBytes, stream>>>(
            w_hh, h0, out, (unsigned*)d_ws, nullptr, nullptr);
    }
}
